// HybridModel_GF_Ratio_6073083757117
// MI455X (gfx1250) — matrix operand order repaired and checked by reading, not compiled, not run on hardware
//
#include <hip/hip_runtime.h>
#include <hip/hip_bf16.h>
#include <math.h>

typedef __attribute__((ext_vector_type(16))) __bf16 v16bf;
typedef __attribute__((ext_vector_type(8)))  __bf16 v8bf;
typedef __attribute__((ext_vector_type(8)))  float  v8f;

#define VOCAB  32000
#define TSEQ   1024
#define DMODEL 1024
#define NLAYER 4
#define NHEAD  16
#define HEAD   64
#define DIN    2048
#define MHEADN 32
#define DSTATE 64
#define FFH    4096
#define ZCOLS  4256   // 2*DIN + 2*DSTATE + MHEAD
#define XBCC   2176   // DIN + 2*DSTATE
#define EPS    1e-6f

#define LDSTRIDE 40   // 80B row stride: keeps all fragment reads 16B-aligned

#if defined(__gfx1250__) && __has_builtin(__builtin_amdgcn_global_load_async_to_lds_b32)
#define HAVE_ASYNC_LDS 1
typedef __attribute__((address_space(1))) int GASI;   // global (AS1)
typedef __attribute__((address_space(3))) int LDSI;   // LDS (AS3)
__device__ __forceinline__ void async_wait_all() {
#if __has_builtin(__builtin_amdgcn_s_wait_asynccnt)
    __builtin_amdgcn_s_wait_asynccnt(0);
#else
    asm volatile("s_wait_asynccnt 0" ::: "memory");
#endif
}
#else
#define HAVE_ASYNC_LDS 0
#endif

// ---------------------------------------------------------------------------
// bf16 WMMA GEMM: C[M,N] = A[M,K] x B[K,N] (+bias) (+=C), fp32 in/out.
// 128x128 workgroup tile, 8 waves (2x4), each wave 64x32 = 4x2 WMMA tiles.
// Double-buffered LDS; A tile m-major, B tile n-major (k contiguous) so every
// fragment is two aligned ds_load_b128 + shufflevector. All fragments are
// preloaded before the WMMA burst so one s_wait_dscnt covers the whole chunk.
// ---------------------------------------------------------------------------
__device__ __forceinline__ v16bf frag16(const __bf16* p0, const __bf16* p1)
{
    v8bf lo = *(const v8bf*)p0;
    v8bf hi = *(const v8bf*)p1;
    return __builtin_shufflevector(lo, hi, 0,1,2,3,4,5,6,7,8,9,10,11,12,13,14,15);
}

__device__ __forceinline__ void stage_tiles(
    __bf16 (*As)[LDSTRIDE], __bf16 (*Bt)[LDSTRIDE],
    const float* __restrict__ A, int lda,
    const float* __restrict__ Bsrc, int ldb,
    int rowBase, int colBase, int kk, int N, int transB, int tid)
{
    // A tile: 128 rows x 32 k, fully vectorized both sides
    {
        const int ar = tid >> 1, ac0 = (tid & 1) * 16;
        const float* ap = A + (size_t)(rowBase + ar) * lda + kk + ac0;
        v8bf w0, w1;
#pragma unroll
        for (int i = 0; i < 8; ++i) { w0[i] = (__bf16)ap[i]; w1[i] = (__bf16)ap[8 + i]; }
        *(v8bf*)&As[ar][ac0]     = w0;
        *(v8bf*)&As[ar][ac0 + 8] = w1;
    }
    if (transB) {
        // Bsrc is (N x K) row-major: k contiguous per n -> fully vectorized
        const int n = tid >> 1, kb0 = (tid & 1) * 16;
        const int gn = colBase + n;
        v8bf w0, w1;
        if (gn < N) {
            const float* bp = Bsrc + (size_t)gn * ldb + kk + kb0;
#pragma unroll
            for (int i = 0; i < 8; ++i) { w0[i] = (__bf16)bp[i]; w1[i] = (__bf16)bp[8 + i]; }
        } else {
#pragma unroll
            for (int i = 0; i < 8; ++i) { w0[i] = (__bf16)0.f; w1[i] = (__bf16)0.f; }
        }
        *(v8bf*)&Bt[n][kb0]     = w0;
        *(v8bf*)&Bt[n][kb0 + 8] = w1;
    } else {
        // Bsrc is (K x N) row-major: coalesced reads over n, scatter-store k
        const int k = tid >> 3, n0 = (tid & 7) * 16;
        const float* bp = Bsrc + (size_t)(kk + k) * ldb + colBase + n0;
#pragma unroll
        for (int i = 0; i < 16; ++i) {
            const int gn = colBase + n0 + i;
            const float bv = (gn < N) ? bp[i] : 0.f;
            Bt[n0 + i][k] = (__bf16)bv;
        }
    }
}

__device__ __forceinline__ void compute_chunk(
    const __bf16 (*As)[LDSTRIDE], const __bf16 (*Bt)[LDSTRIDE],
    v8f acc[4][2], int wm, int wn, int lane)
{
    const int lm  = lane & 15;
    const int khA = (lane < 16) ? 0 : 8;    // A frag: k runs [khA..+7] and [khA+16..+23]
    const int kbB = (lane < 16) ? 0 : 8;    // B frag: k runs [kbB..+7] and [kbB+16..+23]

    // preload ALL fragments, then issue the 8 WMMAs back-to-back
    v16bf bfr[2];
#pragma unroll
    for (int j = 0; j < 2; ++j) {
        const __bf16* p = &Bt[wn * 32 + j * 16 + lm][kbB];
        bfr[j] = frag16(p, p + 16);
    }
    v16bf afr[4];
#pragma unroll
    for (int i = 0; i < 4; ++i) {
        const __bf16* pa = &As[wm * 64 + i * 16 + lm][khA];
        afr[i] = frag16(pa, pa + 16);
    }
#pragma unroll
    for (int i = 0; i < 4; ++i)
#pragma unroll
        for (int j = 0; j < 2; ++j)
            acc[i][j] = __builtin_amdgcn_wmma_f32_16x16x32_bf16(
                false, afr[i], false, bfr[j], (short)0, acc[i][j], false, false);
}

__global__ __launch_bounds__(256)
void gemm_wmma(const float* __restrict__ A, int lda,
               const float* __restrict__ Bsrc, int ldb,
               float* __restrict__ C, int ldc,
               int M, int N, int K, int transB,
               const float* __restrict__ bias, int addC)
{
    __shared__ __attribute__((aligned(16))) __bf16 As[2][128][LDSTRIDE];
    __shared__ __attribute__((aligned(16))) __bf16 Bt[2][128][LDSTRIDE];

    const int tid  = threadIdx.x;
    const int wave = tid >> 5;
    const int lane = tid & 31;
    const int wm   = wave >> 2;   // 0..1
    const int wn   = wave & 3;    // 0..3
    const int rowBase = blockIdx.y * 128;
    const int colBase = blockIdx.x * 128;

    v8f acc[4][2];
    const v8f zacc = {0.f,0.f,0.f,0.f,0.f,0.f,0.f,0.f};
#pragma unroll
    for (int i = 0; i < 4; ++i)
#pragma unroll
        for (int j = 0; j < 2; ++j) acc[i][j] = zacc;

    stage_tiles(As[0], Bt[0], A, lda, Bsrc, ldb, rowBase, colBase, 0, N, transB, tid);
    __syncthreads();

    for (int kk = 0; kk < K; kk += 32) {
        const int cur = (kk >> 5) & 1;
        if (kk + 32 < K) {
            stage_tiles(As[cur ^ 1], Bt[cur ^ 1], A, lda, Bsrc, ldb,
                        rowBase, colBase, kk + 32, N, transB, tid);
            if (kk + 64 < K) {
                __builtin_prefetch(A + (size_t)(rowBase + (tid >> 1)) * lda + kk + 64, 0, 1);
                if (!transB)
                    __builtin_prefetch(Bsrc + (size_t)(kk + 64 + (tid >> 3)) * ldb + colBase + (tid & 7) * 16, 0, 1);
                else
                    __builtin_prefetch(Bsrc + (size_t)(colBase + (tid >> 1)) * ldb + kk + 64, 0, 1);
            }
        }
        compute_chunk(As[cur], Bt[cur], acc, wm, wn, lane);
        __syncthreads();
    }

    // epilogue: D layout = lanes 0-15: N=lane, M=v ; lanes 16-31: N=lane-16, M=8+v
    const int lm    = lane & 15;
    const int mhalf = (lane < 16) ? 0 : 8;
#pragma unroll
    for (int i = 0; i < 4; ++i) {
#pragma unroll
        for (int j = 0; j < 2; ++j) {
            const int gcol = colBase + wn * 32 + j * 16 + lm;
            if (gcol < N) {
                const float badd = bias ? bias[gcol] : 0.f;
#pragma unroll
                for (int v2 = 0; v2 < 8; ++v2) {
                    const int grow = rowBase + wm * 64 + i * 16 + mhalf + v2;
                    const size_t o = (size_t)grow * ldc + gcol;
                    float val = acc[i][j][v2] + badd;
                    if (addC) val += C[o];
                    C[o] = val;
                }
            }
        }
    }
}

// ---------------------------------------------------------------------------
// Elementwise / norm / glue kernels
// ---------------------------------------------------------------------------
__global__ void embed_gather(const int* __restrict__ x,
                             const float* __restrict__ embed,
                             float* __restrict__ h)
{
    int idx = blockIdx.x * blockDim.x + threadIdx.x;
    int t = idx >> 10, d = idx & 1023;
    h[idx] = embed[(size_t)x[t] * DMODEL + d];
}

__global__ __launch_bounds__(256)
void rmsnorm_k(const float* __restrict__ in, const float* __restrict__ w,
               float* __restrict__ out, int D)
{
    const int row = blockIdx.x, tid = threadIdx.x;
    __shared__ float red[256];
    const float* ip = in + (size_t)row * D;
    float s = 0.f;
    for (int d = tid; d < D; d += 256) { float v = ip[d]; s += v * v; }
    red[tid] = s; __syncthreads();
    for (int st = 128; st > 0; st >>= 1) {
        if (tid < st) red[tid] += red[tid + st];
        __syncthreads();
    }
    const float inv = rsqrtf(red[0] / (float)D + EPS);
    float* op = out + (size_t)row * D;
    for (int d = tid; d < D; d += 256) op[d] = w[d] * ip[d] * inv;
}

__global__ void ew_decay(float* p, int n) {           // exp(-exp(x))
    int i = blockIdx.x * blockDim.x + threadIdx.x;
    if (i < n) p[i] = __expf(-__expf(p[i]));
}
__global__ void ew_silu(float* p, int n) {
    int i = blockIdx.x * blockDim.x + threadIdx.x;
    if (i < n) { float v = p[i]; p[i] = v / (1.f + __expf(-v)); }
}
__global__ void ew_mul(const float* a, const float* b, float* o, int n) {
    int i = blockIdx.x * blockDim.x + threadIdx.x;
    if (i < n) o[i] = a[i] * b[i];
}
__global__ void ew_gelu(float* p, int n) {            // exact gelu
    int i = blockIdx.x * blockDim.x + threadIdx.x;
    if (i < n) { float v = p[i]; p[i] = 0.5f * v * (1.f + erff(v * 0.70710678118f)); }
}

// causal depthwise conv (K=4) + silu over xBC slice of zxbcdt
__global__ void conv_silu(const float* __restrict__ zx,
                          const float* __restrict__ cw,
                          float* __restrict__ out)
{
    int idx = blockIdx.x * blockDim.x + threadIdx.x;
    if (idx >= TSEQ * XBCC) return;
    int t = idx / XBCC, c = idx % XBCC;
    float s = 0.f;
#pragma unroll
    for (int j = 0; j < 4; ++j) {
        int tt = t - 3 + j;
        if (tt >= 0) s += zx[(size_t)tt * ZCOLS + DIN + c] * cw[c * 4 + j];
    }
    out[idx] = s / (1.f + __expf(-s));
}

__global__ void dt_da_k(const float* __restrict__ zx,
                        const float* __restrict__ dtb_p,
                        const float* __restrict__ alog,
                        float* __restrict__ dtb, float* __restrict__ dab)
{
    int idx = blockIdx.x * blockDim.x + threadIdx.x;
    if (idx >= TSEQ * MHEADN) return;
    int t = idx / MHEADN, hm = idx % MHEADN;
    float v = zx[(size_t)t * ZCOLS + (ZCOLS - MHEADN) + hm] + dtb_p[hm];
    float sp = (v > 20.f) ? v : log1pf(__expf(v));
    dtb[idx] = sp;
    dab[idx] = __expf(-sp * __expf(alog[hm]));
}

// RWKV recurrence: block per head, state S[64k x 64v] in registers.
// Per-step r/k/v/dec vectors are pure global->LDS copies: use async LDS loads.
__global__ __launch_bounds__(256)
void rwkv_scan(const float* __restrict__ rB, const float* __restrict__ kB,
               const float* __restrict__ vB, const float* __restrict__ dB,
               const float* __restrict__ u, float* __restrict__ oB)
{
    const int head = blockIdx.x, tid = threadIdx.x;
    const int vi = tid & 63, kg = tid >> 6;   // 4 k-groups of 16
    float S[16];
#pragma unroll
    for (int i = 0; i < 16; ++i) S[i] = 0.f;

    __shared__ float rl[64], kl[64], vl[64], dl[64], ul[64];
    __shared__ float red[4][64];
    if (tid < 64) ul[tid] = u[head * 64 + tid];
    __syncthreads();

    for (int t = 0; t < TSEQ; ++t) {
        const int base = t * DMODEL + head * 64;
#if HAVE_ASYNC_LDS
        if (tid < 64) {
            __builtin_amdgcn_global_load_async_to_lds_b32((GASI*)(rB + base + tid), (LDSI*)&rl[tid], 0, 0);
            __builtin_amdgcn_global_load_async_to_lds_b32((GASI*)(kB + base + tid), (LDSI*)&kl[tid], 0, 0);
            __builtin_amdgcn_global_load_async_to_lds_b32((GASI*)(vB + base + tid), (LDSI*)&vl[tid], 0, 0);
            __builtin_amdgcn_global_load_async_to_lds_b32((GASI*)(dB + base + tid), (LDSI*)&dl[tid], 0, 0);
        }
        async_wait_all();
#else
        if (tid < 64) {
            rl[tid] = rB[base + tid]; kl[tid] = kB[base + tid];
            vl[tid] = vB[base + tid]; dl[tid] = dB[base + tid];
        }
#endif
        __syncthreads();
        const float vv = vl[vi];
        float p = 0.f;
#pragma unroll
        for (int i = 0; i < 16; ++i) {
            const int ki = kg * 16 + i;
            const float kv = kl[ki] * vv;
            p += rl[ki] * (S[i] + ul[ki] * kv);
            S[i] = dl[ki] * S[i] + kv;
        }
        red[kg][vi] = p;
        __syncthreads();
        if (kg == 0)
            oB[base + vi] = red[0][vi] + red[1][vi] + red[2][vi] + red[3][vi];
        __syncthreads();
    }
}

// Mamba SSM recurrence: block per mhead, state h[64p x 64s] in registers
__global__ __launch_bounds__(256)
void mamba_scan(const float* __restrict__ xc, const float* __restrict__ dtb,
                const float* __restrict__ dab, const float* __restrict__ Dsk,
                float* __restrict__ ym)
{
    const int head = blockIdx.x, tid = threadIdx.x;
    const int p = tid & 63, sg = tid >> 6;
    float S[16];
#pragma unroll
    for (int i = 0; i < 16; ++i) S[i] = 0.f;

    __shared__ float xl[64], Bl[64], Cl[64];
    __shared__ float red[4][64];
    __shared__ float sc[2];
    const float Dv = Dsk[head];

    for (int t = 0; t < TSEQ; ++t) {
#if HAVE_ASYNC_LDS
        if (tid < 64) {
            __builtin_amdgcn_global_load_async_to_lds_b32((GASI*)(xc + (size_t)t * XBCC + head * 64 + tid), (LDSI*)&xl[tid], 0, 0);
            __builtin_amdgcn_global_load_async_to_lds_b32((GASI*)(xc + (size_t)t * XBCC + DIN + tid),          (LDSI*)&Bl[tid], 0, 0);
            __builtin_amdgcn_global_load_async_to_lds_b32((GASI*)(xc + (size_t)t * XBCC + DIN + DSTATE + tid), (LDSI*)&Cl[tid], 0, 0);
        }
        if (tid == 64) { sc[0] = dab[t * MHEADN + head]; sc[1] = dtb[t * MHEADN + head]; }
        async_wait_all();
#else
        if (tid < 64) {
            xl[tid] = xc[(size_t)t * XBCC + head * 64 + tid];
            Bl[tid] = xc[(size_t)t * XBCC + DIN + tid];
            Cl[tid] = xc[(size_t)t * XBCC + DIN + DSTATE + tid];
        }
        if (tid == 64) { sc[0] = dab[t * MHEADN + head]; sc[1] = dtb[t * MHEADN + head]; }
#endif
        __syncthreads();
        const float dA = sc[0], dtv = sc[1];
        const float xp = xl[p], coef = dtv * xp;
        float accv = 0.f;
#pragma unroll
        for (int i = 0; i < 16; ++i) {
            const int s = sg * 16 + i;
            S[i] = dA * S[i] + coef * Bl[s];
            accv += S[i] * Cl[s];
        }
        red[sg][p] = accv;
        __syncthreads();
        if (sg == 0)
            ym[(size_t)t * DIN + head * 64 + p] =
                red[0][p] + red[1][p] + red[2][p] + red[3][p] + Dv * xp;
        __syncthreads();
    }
}

// ym * silu(z) followed by rmsnorm with mnorm_w (D = 2048)
__global__ __launch_bounds__(256)
void msilu_rmsnorm(const float* __restrict__ ym, const float* __restrict__ zx,
                   const float* __restrict__ mw, float* __restrict__ out)
{
    const int row = blockIdx.x, tid = threadIdx.x;
    __shared__ float vals[DIN];
    __shared__ float red[256];
    float s = 0.f;
    for (int d = tid; d < DIN; d += 256) {
        float z = zx[(size_t)row * ZCOLS + d];
        float v = ym[(size_t)row * DIN + d] * (z / (1.f + __expf(-z)));
        vals[d] = v; s += v * v;
    }
    red[tid] = s; __syncthreads();
    for (int st = 128; st > 0; st >>= 1) {
        if (tid < st) red[tid] += red[tid + st];
        __syncthreads();
    }
    const float inv = rsqrtf(red[0] / (float)DIN + EPS);
    for (int d = tid; d < DIN; d += 256)
        out[(size_t)row * DIN + d] = mw[d] * vals[d] * inv;
}

// sigmoid gate over concat dot, then residual mix into h
__global__ __launch_bounds__(256)
void gate_residual(float* __restrict__ h, const float* __restrict__ rw,
                   const float* __restrict__ mb, const float* __restrict__ gw,
                   const float* __restrict__ gbp)
{
    const int row = blockIdx.x, tid = threadIdx.x;
    __shared__ float red[256];
    float s = 0.f;
    for (int d = tid; d < DMODEL; d += 256)
        s += rw[(size_t)row * DMODEL + d] * gw[d] +
             mb[(size_t)row * DMODEL + d] * gw[DMODEL + d];
    red[tid] = s; __syncthreads();
    for (int st = 128; st > 0; st >>= 1) {
        if (tid < st) red[tid] += red[tid + st];
        __syncthreads();
    }
    const float gate = 1.f / (1.f + __expf(-(red[0] + gbp[0])));
    for (int d = tid; d < DMODEL; d += 256) {
        const size_t o = (size_t)row * DMODEL + d;
        h[o] += gate * rw[o] + (1.f - gate) * mb[o];
    }
}

// ---------------------------------------------------------------------------
static void gemm_launch(const float* A, int lda, const float* B, int ldb,
                        float* C, int ldc, int M, int N, int K, int transB,
                        const float* bias, int addC, hipStream_t s)
{
    dim3 grid((N + 127) / 128, M / 128);
    gemm_wmma<<<grid, 256, 0, s>>>(A, lda, B, ldb, C, ldc, M, N, K, transB, bias, addC);
}

extern "C" void kernel_launch(void* const* d_in, const int* in_sizes, int n_in,
                              void* d_out, int out_size, void* d_ws, size_t ws_size,
                              hipStream_t stream)
{
    const int*   x      = (const int*)  d_in[0];
    const float* embed  = (const float*)d_in[1];
    const float* ln_w   = (const float*)d_in[2];
    const float* Wr     = (const float*)d_in[3];
    const float* Wk     = (const float*)d_in[4];
    const float* Wv     = (const float*)d_in[5];
    const float* Ww     = (const float*)d_in[6];
    const float* w_bias = (const float*)d_in[7];
    const float* u      = (const float*)d_in[8];
    const float* Wg     = (const float*)d_in[9];
    const float* Wo     = (const float*)d_in[10];
    const float* Win    = (const float*)d_in[11];
    const float* conv_w = (const float*)d_in[12];
    const float* dt_bias= (const float*)d_in[13];
    const float* A_log  = (const float*)d_in[14];
    const float* Dskip  = (const float*)d_in[15];
    const float* mnorm_w= (const float*)d_in[16];
    const float* Wout_m = (const float*)d_in[17];
    const float* gate_w = (const float*)d_in[18];
    const float* gate_b = (const float*)d_in[19];
    const float* ffn_lnw= (const float*)d_in[20];
    const float* W1     = (const float*)d_in[21];
    const float* W2     = (const float*)d_in[22];
    const float* ln_out = (const float*)d_in[23];
    float* logits = (float*)d_out;

    // workspace bump allocator
    float* ws = (float*)d_ws;
    size_t off = 0;
    auto alloc = [&](size_t n) { float* p = ws + off; off += n; return p; };
    const size_t TD = (size_t)TSEQ * DMODEL;
    float* h        = alloc(TD);
    float* nx       = alloc(TD);
    float* rB       = alloc(TD);
    float* kB       = alloc(TD);
    float* vB       = alloc(TD);
    float* decB     = alloc(TD);
    float* gB       = alloc(TD);
    float* oB       = alloc(TD);
    float* og       = alloc(TD);
    float* out_rwkv = alloc(TD);
    float* zx       = alloc((size_t)TSEQ * ZCOLS);
    float* xBCc     = alloc((size_t)TSEQ * XBCC);
    float* dtB      = alloc((size_t)TSEQ * MHEADN);
    float* dAB      = alloc((size_t)TSEQ * MHEADN);
    float* ymB      = alloc((size_t)TSEQ * DIN);
    float* ymn      = alloc((size_t)TSEQ * DIN);
    float* out_mam  = alloc(TD);
    float* nx2      = alloc(TD);
    float* ffh      = alloc((size_t)TSEQ * FFH);
    float* nf       = alloc(TD);
    (void)ws_size; (void)n_in; (void)in_sizes; (void)out_size;

    embed_gather<<<(TSEQ * DMODEL) / 256, 256, 0, stream>>>(x, embed, h);

    for (int l = 0; l < NLAYER; ++l) {
        const float* ln_l  = ln_w    + (size_t)l * DMODEL;
        const float* Wr_l  = Wr      + (size_t)l * DMODEL * DMODEL;
        const float* Wk_l  = Wk      + (size_t)l * DMODEL * DMODEL;
        const float* Wv_l  = Wv      + (size_t)l * DMODEL * DMODEL;
        const float* Ww_l  = Ww      + (size_t)l * DMODEL * DMODEL;
        const float* wb_l  = w_bias  + (size_t)l * DMODEL;
        const float* u_l   = u       + (size_t)l * NHEAD * HEAD;
        const float* Wg_l  = Wg      + (size_t)l * DMODEL * DMODEL;
        const float* Wo_l  = Wo      + (size_t)l * DMODEL * DMODEL;
        const float* Win_l = Win     + (size_t)l * DMODEL * ZCOLS;
        const float* cw_l  = conv_w  + (size_t)l * XBCC * 4;
        const float* dtb_l = dt_bias + (size_t)l * MHEADN;
        const float* al_l  = A_log   + (size_t)l * MHEADN;
        const float* D_l   = Dskip   + (size_t)l * MHEADN;
        const float* mw_l  = mnorm_w + (size_t)l * DIN;
        const float* Wom_l = Wout_m  + (size_t)l * DIN * DMODEL;
        const float* gw_l  = gate_w  + (size_t)l * 2 * DMODEL;
        const float* gb_l  = gate_b  + l;
        const float* fln_l = ffn_lnw + (size_t)l * DMODEL;
        const float* W1_l  = W1      + (size_t)l * DMODEL * FFH;
        const float* W2_l  = W2      + (size_t)l * FFH * DMODEL;

        // pre-norm + projections
        rmsnorm_k<<<TSEQ, 256, 0, stream>>>(h, ln_l, nx, DMODEL);
        gemm_launch(nx, DMODEL, Wr_l,  DMODEL, rB,   DMODEL, TSEQ, DMODEL, DMODEL, 0, nullptr, 0, stream);
        gemm_launch(nx, DMODEL, Wk_l,  DMODEL, kB,   DMODEL, TSEQ, DMODEL, DMODEL, 0, nullptr, 0, stream);
        gemm_launch(nx, DMODEL, Wv_l,  DMODEL, vB,   DMODEL, TSEQ, DMODEL, DMODEL, 0, nullptr, 0, stream);
        gemm_launch(nx, DMODEL, Ww_l,  DMODEL, decB, DMODEL, TSEQ, DMODEL, DMODEL, 0, wb_l,    0, stream);
        gemm_launch(nx, DMODEL, Wg_l,  DMODEL, gB,   DMODEL, TSEQ, DMODEL, DMODEL, 0, nullptr, 0, stream);
        gemm_launch(nx, DMODEL, Win_l, ZCOLS,  zx,   ZCOLS,  TSEQ, ZCOLS,  DMODEL, 0, nullptr, 0, stream);

        // RWKV branch
        ew_decay<<<(TSEQ * DMODEL + 255) / 256, 256, 0, stream>>>(decB, TSEQ * DMODEL);
        ew_silu <<<(TSEQ * DMODEL + 255) / 256, 256, 0, stream>>>(gB,   TSEQ * DMODEL);
        rwkv_scan<<<NHEAD, 256, 0, stream>>>(rB, kB, vB, decB, u_l, oB);
        ew_mul  <<<(TSEQ * DMODEL + 255) / 256, 256, 0, stream>>>(oB, gB, og, TSEQ * DMODEL);
        gemm_launch(og, DMODEL, Wo_l, DMODEL, out_rwkv, DMODEL, TSEQ, DMODEL, DMODEL, 0, nullptr, 0, stream);

        // Mamba branch
        conv_silu<<<(TSEQ * XBCC + 255) / 256, 256, 0, stream>>>(zx, cw_l, xBCc);
        dt_da_k  <<<(TSEQ * MHEADN + 255) / 256, 256, 0, stream>>>(zx, dtb_l, al_l, dtB, dAB);
        mamba_scan<<<MHEADN, 256, 0, stream>>>(xBCc, dtB, dAB, D_l, ymB);
        msilu_rmsnorm<<<TSEQ, 256, 0, stream>>>(ymB, zx, mw_l, ymn);
        gemm_launch(ymn, DIN, Wom_l, DMODEL, out_mam, DMODEL, TSEQ, DMODEL, DIN, 0, nullptr, 0, stream);

        // gated residual merge
        gate_residual<<<TSEQ, 256, 0, stream>>>(h, out_rwkv, out_mam, gw_l, gb_l);

        // FFN
        rmsnorm_k<<<TSEQ, 256, 0, stream>>>(h, fln_l, nx2, DMODEL);
        gemm_launch(nx2, DMODEL, W1_l, FFH, ffh, FFH, TSEQ, FFH, DMODEL, 0, nullptr, 0, stream);
        ew_gelu<<<(TSEQ * FFH + 255) / 256, 256, 0, stream>>>(ffh, TSEQ * FFH);
        gemm_launch(ffh, FFH, W2_l, DMODEL, h, DMODEL, TSEQ, DMODEL, FFH, 0, nullptr, 1, stream);
    }

    // final norm + logits = nf @ embed^T  (N = 32000, transB)
    rmsnorm_k<<<TSEQ, 256, 0, stream>>>(h, ln_out, nf, DMODEL);
    gemm_launch(nf, DMODEL, embed, DMODEL, logits, VOCAB, TSEQ, VOCAB, DMODEL, 1, nullptr, 0, stream);
}
